// FeatureAlign_17755394801766
// MI455X (gfx1250) — matrix operand order repaired and checked by reading, not compiled, not run on hardware
//
#include <hip/hip_runtime.h>

// Problem constants
#define Bn 8
#define Cn 256
#define On 256
#define Hn 64
#define Wn 64
#define HW 4096            // Hn*Wn
#define DG 4
#define CG 64              // Cn/DG
#define K2 9
#define KTOT 2304          // DG*K2*CG
#define NTOT 32768         // Bn*HW
#define GN_GROUPS 32
#define CH_PER_G 8
#define KCHUNKS 72         // KTOT / 32
#define MSUBS 16           // On / 16

typedef __attribute__((ext_vector_type(16))) __bf16 bf16x16;
typedef __attribute__((ext_vector_type(8)))  float  f32x8;

union BfFrag {
    bf16x16      v;
    unsigned int u[8];
    uint4        q[2];
};

__device__ __forceinline__ unsigned short f32_to_bf16(float f) {
    unsigned int u = __builtin_bit_cast(unsigned int, f);
    u = (u + 0x7FFFu + ((u >> 16) & 1u)) >> 16;   // round-to-nearest-even
    return (unsigned short)u;
}

// ---------------------------------------------------------------------------
// Kernel 1: offset = einsum('oc,bchw->bohw', w_offset[72,4], x_off[8,4,H,W])
// Layout of offs: [b][j][hw], j = (g*9+k)*2 + {0:y,1:x}
// ---------------------------------------------------------------------------
__global__ __launch_bounds__(256) void offset_kernel(
    const float* __restrict__ x_off, const float* __restrict__ w_off,
    float* __restrict__ offs)
{
    unsigned int idx = blockIdx.x * 256u + threadIdx.x;   // < 8*72*4096
    int hw = idx & 4095;
    int bj = idx >> 12;          // b*72 + j
    int j  = bj % 72;
    int b  = bj / 72;
    const float* xb = x_off + (size_t)b * 4 * HW + hw;
    float s = w_off[j*4+0] * xb[0]
            + w_off[j*4+1] * xb[HW]
            + w_off[j*4+2] * xb[2*HW]
            + w_off[j*4+3] * xb[3*HW];
    offs[idx] = s;
}

// ---------------------------------------------------------------------------
// Kernel 2: weights in WMMA-A *fragment order* (bf16):
//   Wt[((it*16 + msub)*32 + lane)*16 + e]
// so each lane's 16x32 A-fragment for k-chunk `it`, O-subtile `msub` is a
// contiguous 32B run -> two global_load_b128 straight into WMMA sources.
// Fragment K-map (ISA 7.12.2, 16-bit A 16x32): element pair i -> K = 2i +
// (i>=4 ? 8:0) + (lane>=16 ? 8:0); M = lane&15.
// Reduction-index order: r = (g*9+k)*64 + c.
// ---------------------------------------------------------------------------
__global__ __launch_bounds__(256) void wprep_kernel(
    const float* __restrict__ wd, unsigned short* __restrict__ Wt)
{
    unsigned int idx = blockIdx.x * 256u + threadIdx.x;   // < 72*16*32*16 = 589824
    int e    = idx & 15;
    int lane = (idx >> 4) & 31;
    int msub = (idx >> 9) & 15;
    int it   = idx >> 13;         // 0..71
    int i    = e >> 1;
    int K    = 2 * i + ((i >= 4) ? 8 : 0) + ((lane >= 16) ? 8 : 0) + (e & 1);
    int r    = it * 32 + K;       // reduction index
    int o    = msub * 16 + (lane & 15);
    int c    = r & 63;
    int gk   = r >> 6;
    int g    = gk / 9;
    int k    = gk - g * 9;
    float w = wd[(size_t)o * KTOT + (g * 64 + c) * 9 + k];
    Wt[idx] = f32_to_bf16(w);
}

// ---------------------------------------------------------------------------
// Kernel 3: implicit-GEMM deformable conv via WMMA bf16.
//   M = O (256), N = B*H*W (32768), K = 2304.  Tile: BM=128, BN=32, Kc=32.
//   256 threads = 8 waves; wave w owns M-subtile w, both N-subtiles.
//   A fragments: direct global b128 loads (fragment-ordered Wt, L2-resident).
//   B tile: cooperatively bilinear-sampled into LDS (shared by all 8 waves).
// ---------------------------------------------------------------------------
__global__ __launch_bounds__(256) void deform_gemm_kernel(
    const float* __restrict__ x, const float* __restrict__ offs,
    const unsigned short* __restrict__ Wt, float* __restrict__ y)
{
    __shared__ unsigned short Bs[32][36];    // [n][k], padded row -> no bank conflicts

    const int t    = threadIdx.x;
    const int lane = t & 31;
    const int wave = t >> 5;
    const int mBase = blockIdx.y * 128;
    const int nBase = blockIdx.x * 32;
    const int msubG = blockIdx.y * 8 + wave;   // global O-subtile of this wave

    f32x8 acc0 = {};
    f32x8 acc1 = {};

    // B-fill mapping: 32 positions x 32 channels, 4 channels per thread
    const int nF  = t >> 3;            // 0..31
    const int cF  = (t & 7) * 4;       // 0..28
    const int ngF = nBase + nF;
    const int bF  = ngF >> 12;
    const int hwF = ngF & 4095;
    const int hF  = hwF >> 6;
    const int wF  = hwF & 63;

    for (int it = 0; it < KCHUNKS; ++it) {
        const int kBase = it * 32;
        const int gk  = kBase >> 6;       // fixed (g,k) for the whole chunk
        const int c0  = kBase & 63;       // 0 or 32
        const int g   = gk / 9;
        const int k9  = gk - g * 9;
        const float kyf = (float)(k9 / 3 - 1);
        const float kxf = (float)(k9 % 3 - 1);

        __syncthreads();

        // ---- A fragment: two coalesced 16B global loads, no LDS ----
        BfFrag a;
        {
            const unsigned short* afp =
                Wt + ((size_t)(it * MSUBS + msubG) * 32 + lane) * 16;
            a.q[0] = *(const uint4*)(afp);
            a.q[1] = *(const uint4*)(afp + 8);
        }

        // ---- fill B tile: bilinear sample 4 channels at (nF, g, k9) ----
        {
            const size_t obase = (size_t)bF * (72 * HW) + (size_t)((g * 9 + k9) * 2) * HW + hwF;
            const float offy = offs[obase];
            const float offx = offs[obase + HW];
            float py = (float)hF + kyf + offy;
            float px = (float)wF + kxf + offx;
            float y0 = floorf(py), x0 = floorf(px);
            float wy = py - y0,    wx = px - x0;
            int y0i = (int)y0, x0i = (int)x0;
            int y1i = y0i + 1,  x1i = x0i + 1;
            float vy0 = (y0i >= 0 && y0i < Hn) ? 1.f : 0.f;
            float vy1 = (y1i >= 0 && y1i < Hn) ? 1.f : 0.f;
            float vx0 = (x0i >= 0 && x0i < Wn) ? 1.f : 0.f;
            float vx1 = (x1i >= 0 && x1i < Wn) ? 1.f : 0.f;
            int y0c = y0i < 0 ? 0 : (y0i > 63 ? 63 : y0i);
            int y1c = y1i < 0 ? 0 : (y1i > 63 ? 63 : y1i);
            int x0c = x0i < 0 ? 0 : (x0i > 63 ? 63 : x0i);
            int x1c = x1i < 0 ? 0 : (x1i > 63 ? 63 : x1i);
            float w00 = (1.f - wy) * (1.f - wx) * vy0 * vx0;
            float w01 = (1.f - wy) * wx        * vy0 * vx1;
            float w10 = wy        * (1.f - wx) * vy1 * vx0;
            float w11 = wy        * wx         * vy1 * vx1;
            int i00 = y0c * 64 + x0c, i01 = y0c * 64 + x1c;
            int i10 = y1c * 64 + x0c, i11 = y1c * 64 + x1c;
            const float* xc = x + ((size_t)bF * Cn + g * CG + c0 + cF) * HW;
#pragma unroll
            for (int j = 0; j < 4; ++j) {
                const float* xp = xc + (size_t)j * HW;
                float v = w00 * xp[i00] + w01 * xp[i01] + w10 * xp[i10] + w11 * xp[i11];
                Bs[nF][cF + j] = f32_to_bf16(v);
            }
        }

        __syncthreads();

        // ---- B fragments (ISA 7.12.2 32x16 bf16 layout) + WMMA ----
        {
            BfFrag b0, b1;
            const int nrow = lane & 15;
            const int hiB  = (lane >= 16) ? 8 : 0;
#pragma unroll
            for (int i = 0; i < 8; ++i) {
                b0.u[i] = *(const unsigned int*)&Bs[nrow][2 * i + ((i >= 4) ? 8 : 0) + hiB];
                b1.u[i] = *(const unsigned int*)&Bs[16 + nrow][2 * i + ((i >= 4) ? 8 : 0) + hiB];
            }
            acc0 = __builtin_amdgcn_wmma_f32_16x16x32_bf16(false, a.v, false, b0.v,
                                                           (short)0, acc0, false, false);
            acc1 = __builtin_amdgcn_wmma_f32_16x16x32_bf16(false, a.v, false, b1.v,
                                                           (short)0, acc1, false, false);
        }
    }

    // ---- store pre-GN y to d_out (C/D layout: VGPR v -> M = v + 8*(lane>=16)) ----
    const int hiD  = (lane >= 16) ? 8 : 0;
    const int ncol = lane & 15;
#pragma unroll
    for (int vv = 0; vv < 8; ++vv) {
        int o = mBase + wave * 16 + vv + hiD;
        {
            int ng = nBase + ncol;
            int b  = ng >> 12, hw = ng & 4095;
            y[((size_t)b * On + o) * HW + hw] = acc0[vv];
        }
        {
            int ng = nBase + 16 + ncol;
            int b  = ng >> 12, hw = ng & 4095;
            y[((size_t)b * On + o) * HW + hw] = acc1[vv];
        }
    }
}

// ---------------------------------------------------------------------------
// Kernel 4: GroupNorm statistics per (b, group): mean, rsqrt(var+eps)
// ---------------------------------------------------------------------------
__global__ __launch_bounds__(256) void gn_stats_kernel(
    const float* __restrict__ y, float* __restrict__ stats)
{
    __shared__ float s1[256];
    __shared__ float s2[256];
    const int bg = blockIdx.x;                       // b*32 + grp, 0..255
    const size_t base = (size_t)bg * CH_PER_G * HW;  // (b*256 + grp*8)*4096
    const int t = threadIdx.x;
    float sum = 0.f, sq = 0.f;
    for (int i = t; i < CH_PER_G * HW; i += 256) {
        float v = y[base + i];
        sum += v;
        sq  += v * v;
    }
    s1[t] = sum; s2[t] = sq;
    __syncthreads();
    for (int off = 128; off > 0; off >>= 1) {
        if (t < off) { s1[t] += s1[t + off]; s2[t] += s2[t + off]; }
        __syncthreads();
    }
    if (t == 0) {
        const float inv_n = 1.f / (float)(CH_PER_G * HW);
        float mu  = s1[0] * inv_n;
        float var = s2[0] * inv_n - mu * mu;
        stats[bg * 2 + 0] = mu;
        stats[bg * 2 + 1] = rsqrtf(var + 1e-5f);
    }
}

// ---------------------------------------------------------------------------
// Kernel 5: normalize + affine + ReLU, in place in d_out
// ---------------------------------------------------------------------------
__global__ __launch_bounds__(256) void gn_apply_kernel(
    float* __restrict__ y, const float* __restrict__ stats,
    const float* __restrict__ gamma, const float* __restrict__ beta)
{
    size_t idx = (size_t)blockIdx.x * 256 + threadIdx.x;  // < 8*256*4096
    int o  = (int)((idx >> 12) & 255);
    int b  = (int)(idx >> 20);
    int bg = b * GN_GROUPS + (o >> 3);
    float mu  = stats[bg * 2 + 0];
    float inv = stats[bg * 2 + 1];
    float v = (y[idx] - mu) * inv * gamma[o] + beta[o];
    y[idx] = fmaxf(v, 0.f);
}

// ---------------------------------------------------------------------------
extern "C" void kernel_launch(void* const* d_in, const int* in_sizes, int n_in,
                              void* d_out, int out_size, void* d_ws, size_t ws_size,
                              hipStream_t stream)
{
    const float* x      = (const float*)d_in[0];   // [8,256,64,64]
    const float* x_off  = (const float*)d_in[1];   // [8,4,64,64]
    const float* w_off  = (const float*)d_in[2];   // [72,4]
    const float* w_def  = (const float*)d_in[3];   // [256,256,3,3]
    const float* gamma  = (const float*)d_in[4];   // [256]
    const float* beta   = (const float*)d_in[5];   // [256]
    float* out = (float*)d_out;                    // [8,256,64,64] f32

    // Workspace layout
    const size_t offs_bytes = (size_t)Bn * 72 * HW * sizeof(float);       // 9,437,184
    const size_t wt_bytes   = (size_t)KTOT * On * sizeof(unsigned short); // 1,179,648
    float*          offs  = (float*)d_ws;
    unsigned short* Wt    = (unsigned short*)((char*)d_ws + offs_bytes);
    float*          stats = (float*)((char*)d_ws + offs_bytes + wt_bytes);

    // 1) offsets: 8*72*4096 = 2,359,296 elements
    offset_kernel<<<(Bn * 72 * HW) / 256, 256, 0, stream>>>(x_off, w_off, offs);

    // 2) bf16 weights in fragment order: 589,824 elements
    wprep_kernel<<<(KTOT * On) / 256, 256, 0, stream>>>(w_def, Wt);

    // 3) implicit-GEMM deformable conv: grid = (N/32, O/128)
    dim3 grid(NTOT / 32, On / 128);
    deform_gemm_kernel<<<grid, 256, 0, stream>>>(x, offs, Wt, out);

    // 4) GN stats: one block per (b, group)
    gn_stats_kernel<<<Bn * GN_GROUPS, 256, 0, stream>>>(out, stats);

    // 5) normalize + ReLU in place
    gn_apply_kernel<<<(Bn * On * HW) / 256, 256, 0, stream>>>(out, stats, gamma, beta);
}
